// MultiPoolReadout_63221918597535
// MI455X (gfx1250) — compile-verified, matrix operand order checked by reading, not run on hardware
//
#include <hip/hip_runtime.h>
#include <math.h>

#define NUM_GRAPHS 512
#define N_NODES    200000
#define HIDDEN     256
#define H4         64

typedef __attribute__((ext_vector_type(16))) __bf16 v16bf;
typedef __attribute__((ext_vector_type(8)))  float  v8f;
typedef __attribute__((ext_vector_type(4))) unsigned int u32x4;
typedef __attribute__((ext_vector_type(8))) int i32x8;
typedef __attribute__((ext_vector_type(4))) int i32x4;

#if defined(__gfx1250__) && __has_builtin(__builtin_amdgcn_tensor_load_to_lds)
#define USE_TDM 1
#endif

// ---------------------------------------------------------------------------
// Kernel 1: per-graph start offsets via lower_bound on sorted batch[].
// ---------------------------------------------------------------------------
__global__ void offsets_kernel(const int* __restrict__ batch,
                               int* __restrict__ offs, int n) {
    int t = blockIdx.x * blockDim.x + threadIdx.x;
    if (t > NUM_GRAPHS) return;
    int lo = 0, hi = n;
    while (lo < hi) {
        int mid = (lo + hi) >> 1;
        if (batch[mid] < t) lo = mid + 1; else hi = mid;
    }
    offs[t] = lo;
}

// ---------------------------------------------------------------------------
// Kernel 2: one block (256 thr = 8 waves) per graph.
//   Stage A: gate logits via v_wmma_f32_16x16x32_bf16 (w1 staged by TDM)
//   Stage B: per-channel segment sum & max (fp32), per-graph logit max
//   Stage C: softmax-weighted segment sum (fp32)
// ---------------------------------------------------------------------------
__global__ void __launch_bounds__(256)
per_graph_kernel(const float* __restrict__ x,
                 const int*   __restrict__ offs,
                 const float* __restrict__ w1,   // (256,64) row-major (K,N)
                 const float* __restrict__ b1,   // (64)
                 const float* __restrict__ w2,   // (64)
                 const float* __restrict__ b2,   // (1)
                 float* __restrict__ gbuf,       // (N) gate logits scratch
                 float* __restrict__ zbuf)       // (G,768)
{
    __shared__ __align__(16) float  w1stage[HIDDEN * H4];    // 64 KB TDM target
    __shared__ __align__(32) __bf16 w1frag[8 * 4 * 32 * 16]; // 32 KB fragments
    __shared__ float s_b1[H4];
    __shared__ float s_w2[H4];
    __shared__ float red[256];

    const int g   = blockIdx.x;
    const int tid = threadIdx.x;
    const int n0  = offs[g];
    const int n1  = offs[g + 1];
    const int cnt = n1 - n0;

#ifdef USE_TDM
    // ---- Tensor Data Mover: DMA w1 (256x64 f32, contiguous) into LDS ----
    if (tid == 0) {
        const unsigned long long ga = (unsigned long long)(uintptr_t)w1;
        const unsigned int lds_off = (unsigned int)(uintptr_t)(&w1stage[0]);
        u32x4 g0;
        g0[0] = 1u;                                   // count=1, user desc
        g0[1] = lds_off;                              // lds_addr
        g0[2] = (unsigned int)ga;                     // global_addr[31:0]
        g0[3] = (unsigned int)((ga >> 32) & 0x01FFFFFFu) | (2u << 30); // type=2
        i32x8 g1;
        g1[0] = (int)(2u << 16);                      // data_size=4B, no mask
        g1[1] = (int)(64u << 16);                     // tensor_dim0=64 (lo16)
        g1[2] = (int)(256u << 16);                    // dim0 hi=0 | tensor_dim1=256
        g1[3] = (int)(64u << 16);                     // dim1 hi=0 | tile_dim0=64
        g1[4] = (int)256u;                            // tile_dim1=256, tile_dim2=0
        g1[5] = (int)64u;                             // tensor_dim0_stride=64
        g1[6] = 0;
        g1[7] = 0;
        i32x4 gz = {0, 0, 0, 0};
#if __clang_major__ >= 23
        i32x8 gz8 = {0, 0, 0, 0, 0, 0, 0, 0};
        __builtin_amdgcn_tensor_load_to_lds(g0, g1, gz, gz, gz8, 0);
#else
        __builtin_amdgcn_tensor_load_to_lds(g0, g1, gz, gz, 0);
#endif
        __builtin_amdgcn_s_wait_tensorcnt(0);
    }
    __syncthreads();
#else
    for (int i = tid; i < HIDDEN * H4; i += 256) w1stage[i] = w1[i];
    __syncthreads();
#endif

    // ---- swizzle w1 into exact 16-bit B-fragment order (bf16) ----
    // [kb(8)][nTile(4)][lane(32)][elem(16)]; K = e + 8*(e/8) + 8*half
    for (int idx = tid; idx < 8 * 4 * 32 * 16; idx += 256) {
        const int e    = idx & 15;
        const int lane = (idx >> 4) & 31;
        const int nt   = (idx >> 9) & 3;
        const int kb   = idx >> 11;
        const int hfl  = lane >> 4;
        const int k = kb * 32 + e + 8 * (e >> 3) + 8 * hfl;
        const int n = nt * 16 + (lane & 15);
        w1frag[idx] = (__bf16)w1stage[k * H4 + n];
    }
    if (tid < H4) { s_b1[tid] = b1[tid]; s_w2[tid] = w2[tid]; }
    __syncthreads();

    const float b2v   = b2[0];
    const int   wave  = tid >> 5;
    const int   lane  = tid & 31;
    const int   hf    = lane >> 4;
    const int   lan16 = lane & 15;

    // ---- Stage A: gate logits, 16 nodes per wave iteration ----
    for (int base = n0 + wave * 16; base < n1; base += 8 * 16) {
        // Branch-free: clamp the row; garbage rows only pollute their own
        // C-rows, whose gbuf writes are guarded below.
        const int rowc = min(base + lan16, n1 - 1);
        const float* xrow = x + (size_t)rowc * HIDDEN + 8 * hf;
        v8f acc0 = {}, acc1 = {}, acc2 = {}, acc3 = {};

        #pragma unroll 2
        for (int kb = 0; kb < 8; ++kb) {
            const float* xr = xrow + kb * 32;
            const float4 p0 = *(const float4*)(xr);
            const float4 p1 = *(const float4*)(xr + 4);
            const float4 q0 = *(const float4*)(xr + 16);
            const float4 q1 = *(const float4*)(xr + 20);
            v16bf a;
            a[0]=(__bf16)p0.x; a[1]=(__bf16)p0.y; a[2]=(__bf16)p0.z; a[3]=(__bf16)p0.w;
            a[4]=(__bf16)p1.x; a[5]=(__bf16)p1.y; a[6]=(__bf16)p1.z; a[7]=(__bf16)p1.w;
            a[8]=(__bf16)q0.x; a[9]=(__bf16)q0.y; a[10]=(__bf16)q0.z; a[11]=(__bf16)q0.w;
            a[12]=(__bf16)q1.x; a[13]=(__bf16)q1.y; a[14]=(__bf16)q1.z; a[15]=(__bf16)q1.w;

            const v16bf bb0 = *(const v16bf*)&w1frag[((kb * 4 + 0) * 32 + lane) * 16];
            const v16bf bb1 = *(const v16bf*)&w1frag[((kb * 4 + 1) * 32 + lane) * 16];
            const v16bf bb2 = *(const v16bf*)&w1frag[((kb * 4 + 2) * 32 + lane) * 16];
            const v16bf bb3 = *(const v16bf*)&w1frag[((kb * 4 + 3) * 32 + lane) * 16];
            acc0 = __builtin_amdgcn_wmma_f32_16x16x32_bf16(false, a, false, bb0, (short)0, acc0, false, false);
            acc1 = __builtin_amdgcn_wmma_f32_16x16x32_bf16(false, a, false, bb1, (short)0, acc1, false, false);
            acc2 = __builtin_amdgcn_wmma_f32_16x16x32_bf16(false, a, false, bb2, (short)0, acc2, false, false);
            acc3 = __builtin_amdgcn_wmma_f32_16x16x32_bf16(false, a, false, bb3, (short)0, acc3, false, false);
        }

        // ---- fused layer 2: out[m] = sum_n relu(h[m][n]+b1[n]) * w2[n] ----
        float partial[8];
        #pragma unroll
        for (int j = 0; j < 8; ++j) partial[j] = 0.0f;
        {
            const float w2a = s_w2[lan16],      b1a = s_b1[lan16];
            const float w2b = s_w2[16 + lan16], b1b = s_b1[16 + lan16];
            const float w2c = s_w2[32 + lan16], b1c = s_b1[32 + lan16];
            const float w2d = s_w2[48 + lan16], b1d = s_b1[48 + lan16];
            #pragma unroll
            for (int j = 0; j < 8; ++j) {
                partial[j] = fmaf(fmaxf(acc0[j] + b1a, 0.0f), w2a, partial[j]);
                partial[j] = fmaf(fmaxf(acc1[j] + b1b, 0.0f), w2b, partial[j]);
                partial[j] = fmaf(fmaxf(acc2[j] + b1c, 0.0f), w2c, partial[j]);
                partial[j] = fmaf(fmaxf(acc3[j] + b1d, 0.0f), w2d, partial[j]);
            }
        }
        #pragma unroll
        for (int m = 1; m < 16; m <<= 1) {
            #pragma unroll
            for (int j = 0; j < 8; ++j)
                partial[j] += __shfl_xor(partial[j], m, 32);
        }
        if (lan16 == 0) {
            const int rbase = base + 8 * hf;
            #pragma unroll
            for (int j = 0; j < 8; ++j) {
                const int r = rbase + j;
                if (r < n1) gbuf[r] = partial[j] + b2v;
            }
        }
    }
    __threadfence_block();
    __syncthreads();

    // ---- per-graph logit max ----
    float lm = -INFINITY;
    for (int i = n0 + tid; i < n1; i += 256) lm = fmaxf(lm, gbuf[i]);
    red[tid] = lm;
    __syncthreads();
    #pragma unroll
    for (int s = 128; s > 0; s >>= 1) {
        if (tid < s) red[tid] = fmaxf(red[tid], red[tid + s]);
        __syncthreads();
    }
    const float gmax = (cnt > 0) ? red[0] : 0.0f;
    __syncthreads();

    // ---- Stage B: per-channel sum & max (thread t = channel t) ----
    float ssum = 0.0f, smax = -INFINITY;
    for (int i = n0; i < n1; ++i) {
        if (i + 8 < n1) __builtin_prefetch(x + (size_t)(i + 8) * HIDDEN + tid, 0, 1);
        const float v = x[(size_t)i * HIDDEN + tid];
        ssum += v;
        smax = fmaxf(smax, v);
    }
    const float invc = 1.0f / fmaxf((float)cnt, 1.0f);
    zbuf[(size_t)g * 768 + tid]       = ssum * invc;
    zbuf[(size_t)g * 768 + 256 + tid] = (cnt > 0) ? smax : 0.0f;

    // ---- Stage C: attention pool ----
    float denom = 0.0f, acw = 0.0f;
    for (int i = n0; i < n1; ++i) {
        if (i + 8 < n1) __builtin_prefetch(x + (size_t)(i + 8) * HIDDEN + tid, 0, 1);
        const float e = expf(gbuf[i] - gmax);
        denom += e;
        acw = fmaf(e, x[(size_t)i * HIDDEN + tid], acw);
    }
    zbuf[(size_t)g * 768 + 512 + tid] = acw / fmaxf(denom, 1e-16f);
}

// ---------------------------------------------------------------------------
// Kernel 3: y = z @ wp + bp, then LayerNorm * gamma + beta.  One block/graph.
// ---------------------------------------------------------------------------
__global__ void __launch_bounds__(256)
finalize_kernel(const float* __restrict__ zbuf,
                const float* __restrict__ wp,    // (768,256) row-major
                const float* __restrict__ bp,
                const float* __restrict__ gamma,
                const float* __restrict__ beta,
                float* __restrict__ out)         // (G,256)
{
    __shared__ float zs[768];
    __shared__ float red[256];
    const int g = blockIdx.x, t = threadIdx.x;

    for (int i = t; i < 768; i += 256) zs[i] = zbuf[(size_t)g * 768 + i];
    __syncthreads();

    float acc = bp[t];
    #pragma unroll 8
    for (int k = 0; k < 768; ++k) acc = fmaf(zs[k], wp[k * HIDDEN + t], acc);

    red[t] = acc;
    __syncthreads();
    #pragma unroll
    for (int s = 128; s > 0; s >>= 1) {
        if (t < s) red[t] += red[t + s];
        __syncthreads();
    }
    const float mu = red[0] * (1.0f / HIDDEN);
    __syncthreads();

    const float d = acc - mu;
    red[t] = d * d;
    __syncthreads();
    #pragma unroll
    for (int s = 128; s > 0; s >>= 1) {
        if (t < s) red[t] += red[t + s];
        __syncthreads();
    }
    const float var = red[0] * (1.0f / HIDDEN);

    out[(size_t)g * HIDDEN + t] = d * rsqrtf(var + 1e-5f) * gamma[t] + beta[t];
}

// ---------------------------------------------------------------------------
extern "C" void kernel_launch(void* const* d_in, const int* in_sizes, int n_in,
                              void* d_out, int out_size, void* d_ws, size_t ws_size,
                              hipStream_t stream) {
    const float* x     = (const float*)d_in[0];
    const int*   batch = (const int*)  d_in[1];
    const float* w1    = (const float*)d_in[2];
    const float* b1    = (const float*)d_in[3];
    const float* w2    = (const float*)d_in[4];
    const float* b2    = (const float*)d_in[5];
    const float* wp    = (const float*)d_in[6];
    const float* bp    = (const float*)d_in[7];
    const float* gamma = (const float*)d_in[8];
    const float* beta  = (const float*)d_in[9];
    float* out = (float*)d_out;

    char* ws = (char*)d_ws;
    int*   offs = (int*)ws;                          // 513 * 4 B
    float* gbuf = (float*)(ws + 4096);               // N * 4 B
    float* zbuf = (float*)(ws + 4096 + 803840);      // G * 768 * 4 B

    offsets_kernel<<<3, 256, 0, stream>>>(batch, offs, N_NODES);
    per_graph_kernel<<<NUM_GRAPHS, 256, 0, stream>>>(x, offs, w1, b1, w2, b2,
                                                     gbuf, zbuf);
    finalize_kernel<<<NUM_GRAPHS, 256, 0, stream>>>(zbuf, wp, bp, gamma, beta,
                                                    out);
}
